// CamAttnCon_6124623364433
// MI455X (gfx1250) — matrix operand order repaired and checked by reading, not compiled, not run on hardware
//
#include <hip/hip_runtime.h>
#include <hip/hip_bf16.h>
#include <math.h>

typedef __attribute__((ext_vector_type(16))) _Float16 v16h;
typedef __attribute__((ext_vector_type(8)))  float    v8f;

constexpr int BB = 32, TT = 512, DD = 512, HH = 8, SS = 196;
constexpr int KTOP = 51;     // int(0.1 * 512)
constexpr int LAYER = 2;

// ---------------------------------------------------------------------------
// Kernel 1: cosine-similarity weights via v_wmma_f32_16x16x32_f16.
// Each wave computes 16 rows (t values) of weights for one batch b.
// A tile  = target_embed[b, t0..t0+15, k0..k0+31] (16x32 f16)
// B tile  = fore[b, k0..k0+31] broadcast into all 16 columns (32x16 f16)
// -> D[i][j] = dot(x_row_i, y) for every column j (we read column 0).
// Norms accumulated in f32 from the raw loads (exact-ish), combined across
// the complementary K halves with shfl_xor(16) (wave32).
// ---------------------------------------------------------------------------
__global__ void __launch_bounds__(256) cam_weights_wmma(
    const float* __restrict__ fore,      // (B, D)
    const float* __restrict__ te,        // (B, T, D)
    const int*   __restrict__ targets,   // (B, T+1)
    float* __restrict__ wout)            // (B, T) -> workspace
{
  const int wave  = threadIdx.x >> 5;
  const int lane  = threadIdx.x & 31;
  const int half  = lane >> 4;         // which K-half this lane covers
  const int laneM = lane & 15;         // matrix row within the 16-row tile
  const int b  = blockIdx.x >> 2;
  const int t0 = (blockIdx.x & 3) * 128 + wave * 16;

  const float* xrow = te + ((size_t)b * TT + (t0 + laneM)) * (size_t)DD;
  const float* yv   = fore + (size_t)b * DD;
  const int kofsA = half * 16;         // A: lanes 0-15 -> K 0..15; lanes 16-31 -> K 16..31
  const int kofsB = half * 16;         // B: lanes 0-15 -> K 0..15; lanes 16-31 -> K 16..31

  v8f c = {};
  float xsq = 0.f, ysq = 0.f;

  for (int k0 = 0; k0 < DD; k0 += 32) {
    const float4* pa0 = reinterpret_cast<const float4*>(xrow + k0 + kofsA);
    const float4* pa1 = reinterpret_cast<const float4*>(xrow + k0 + 8 + kofsA);
    const float4* pb  = reinterpret_cast<const float4*>(yv + k0 + kofsB);
    float4 a0 = pa0[0], a1 = pa0[1];
    float4 a2 = pa1[0], a3 = pa1[1];
    float4 b0 = pb[0], b1 = pb[1], b2 = pb[2], b3 = pb[3];

    float av[16] = {a0.x,a0.y,a0.z,a0.w, a1.x,a1.y,a1.z,a1.w,
                    a2.x,a2.y,a2.z,a2.w, a3.x,a3.y,a3.z,a3.w};
    float bv[16] = {b0.x,b0.y,b0.z,b0.w, b1.x,b1.y,b1.z,b1.w,
                    b2.x,b2.y,b2.z,b2.w, b3.x,b3.y,b3.z,b3.w};

    v16h a, bf;
#pragma unroll
    for (int j = 0; j < 16; ++j) {
      xsq += av[j] * av[j];
      ysq += bv[j] * bv[j];
      a[j]  = (_Float16)av[j];
      bf[j] = (_Float16)bv[j];
    }
    // (neg_a, A, neg_b, B, c_mod, C, reuse_a, reuse_b)
    c = __builtin_amdgcn_wmma_f32_16x16x32_f16(false, a, false, bf,
                                               (short)0, c, false, false);
  }

  // combine complementary K halves (row laneM lives in lane and lane^16)
  xsq += __shfl_xor(xsq, 16);
  ysq += __shfl_xor(ysq, 16);

  // D layout: VGPR r, lanes 0-15 -> M=r (N=lane); lanes 16-31 -> M=8+r.
  // Column 0 suffices (all columns identical). Route row j to lane j.
  float num = 0.f;
#pragma unroll
  for (int j = 0; j < 8; ++j) {
    float vlo = __shfl(c[j], 0);   // row j
    float vhi = __shfl(c[j], 16);  // row j+8
    if (lane == j)     num = vlo;
    if (lane == j + 8) num = vhi;
  }

  if (lane < 16) {
    int t = t0 + lane;
    float xn = fmaxf(sqrtf(xsq), 1e-8f);
    float yn = fmaxf(sqrtf(ysq), 1e-8f);
    float w  = num / (xn * yn);
    bool mask = (t == 0) || (targets[(size_t)b * (TT + 1) + t] > 0);
    wout[(size_t)b * TT + t] = mask ? w : -1.0f;
  }
}

// ---------------------------------------------------------------------------
// Kernel 2: per-batch top-k (descending, ties -> smaller index, matching
// jax.lax.top_k), gather relu(w * mean_h attn) only at selected rows,
// average over m, min-max normalize over S.
// ---------------------------------------------------------------------------
__global__ void __launch_bounds__(256) cam_topk_accum(
    const float* __restrict__ attn,      // (L, B, H, T, S)
    const int*   __restrict__ targets,   // (B, T+1)
    const float* __restrict__ win,       // (B, T)
    float* __restrict__ out)             // (B, S)
{
  __shared__ float wsm[TT];
  __shared__ float tot[SS];
  __shared__ float rv[256];
  __shared__ int   ri[256];
  __shared__ float rmn[256];
  __shared__ float rmx[256];
  __shared__ int   scnt;

  const int tid = threadIdx.x;
  const int b   = blockIdx.x;

  if (tid == 0) scnt = 0;
  if (tid < SS) tot[tid] = 0.f;
  __syncthreads();

  int cnt = 0;
#pragma unroll
  for (int i = 0; i < 2; ++i) {
    int t = tid + i * 256;
    wsm[t] = win[(size_t)b * TT + t];
    bool mask = (t == 0) || (targets[(size_t)b * (TT + 1) + t] > 0);
    cnt += mask ? 1 : 0;
  }
  atomicAdd(&scnt, cnt);
  __syncthreads();

  const int m = min((int)ceilf((float)scnt * 0.1f), KTOP);   // >= 1 always
  const float* abase = attn + (((size_t)LAYER * BB + b) * HH) * (size_t)TT * SS;

  for (int i = 0; i < m; ++i) {
    // ---- block argmax, tie -> smaller index ----
    float v0 = wsm[tid], v1 = wsm[tid + 256];
    float bv = v0; int bi = tid;
    if (v1 > bv) { bv = v1; bi = tid + 256; }
    rv[tid] = bv; ri[tid] = bi;
    __syncthreads();
    for (int s = 128; s > 0; s >>= 1) {
      if (tid < s) {
        float ov = rv[tid + s]; int oi = ri[tid + s];
        if (ov > rv[tid] || (ov == rv[tid] && oi < ri[tid])) {
          rv[tid] = ov; ri[tid] = oi;
        }
      }
      __syncthreads();
    }
    int   tb = ri[0];
    float wv = rv[0];
    __syncthreads();

    if (tid == 0) wsm[tb] = -3.0f;   // below any cosine / mask value
    if (tid < SS) {
      float acc = 0.f;
#pragma unroll
      for (int h = 0; h < HH; ++h)
        acc += abase[((size_t)h * TT + tb) * SS + tid];
      tot[tid] += fmaxf(wv * (acc * (1.0f / HH)), 0.0f);
    }
    __syncthreads();
  }

  // ---- normalize: (v - min) / max(max - min, 1e-12) ----
  float v = (tid < SS) ? tot[tid] / (float)m : 0.f;
  rmn[tid] = (tid < SS) ? v :  3.4e38f;
  rmx[tid] = (tid < SS) ? v : -3.4e38f;
  __syncthreads();
  for (int s = 128; s > 0; s >>= 1) {
    if (tid < s) {
      rmn[tid] = fminf(rmn[tid], rmn[tid + s]);
      rmx[tid] = fmaxf(rmx[tid], rmx[tid + s]);
    }
    __syncthreads();
  }
  if (tid < SS) {
    float mn = rmn[0], mx = rmx[0];
    out[(size_t)b * SS + tid] = (v - mn) / fmaxf(mx - mn, 1e-12f);
  }
}

extern "C" void kernel_launch(void* const* d_in, const int* in_sizes, int n_in,
                              void* d_out, int out_size, void* d_ws, size_t ws_size,
                              hipStream_t stream) {
  (void)in_sizes; (void)n_in; (void)out_size; (void)ws_size;
  const float* fore = (const float*)d_in[0];   // (B, D)
  const float* te   = (const float*)d_in[1];   // (B, T, D)
  const float* attn = (const float*)d_in[2];   // (L, B, H, T, S)
  const int*   tgt  = (const int*)d_in[3];     // (B, T+1)
  float* wscr = (float*)d_ws;                  // (B, T) weights scratch
  float* out  = (float*)d_out;                 // (B, S)

  cam_weights_wmma<<<dim3(BB * 4), dim3(256), 0, stream>>>(fore, te, tgt, wscr);
  cam_topk_accum <<<dim3(BB),     dim3(256), 0, stream>>>(attn, tgt, wscr, out);
}
